// LinearChainCRF_82171314307121
// MI455X (gfx1250) — matrix operand order repaired and checked by reading, not compiled, not run on hardware
//
#include <hip/hip_runtime.h>
#include <math.h>

// Linear-chain CRF forward on MI455X (gfx1250, wave32, WMMA).
//
// alpha[b,t,:] = logsumexp_i(alpha[b,t-1,i] + T[i,:]) + em[b,t,:]
// Rescaled into an exp-domain GEMM per step:
//   alpha_new[j] = m + log( sum_i exp(alpha[i]-m) * expT[i,j] ) + em[j]
// v = exp(alpha-m) in f16 (<=1), expT in f16 (held in VGPRs for the whole
// scan), f32 accumulate via v_wmma_f32_16x16x32_f16.  16 sequences per
// block (WMMA M), one 16-label N-tile per wave, K=64 as two chained WMMAs.
// Serial-latency-bound problem: row maxima via shfl_xor trees + registers,
// emissions software-pipelined, raw v_log/v_exp, branchless len-1 capture.

typedef __attribute__((ext_vector_type(16))) _Float16 v16h;
typedef __attribute__((ext_vector_type(8)))  float    v8f;

#define CRF_B   32
#define CRF_T   512
#define CRF_L   64
#define ROWS    16     // sequences per block == WMMA M
#define LPITCH  68     // f32 last-alpha pitch (bank-staggered)
#define VPITCH  88     // f16 v pitch: 176 B/row -> 16B-aligned b128 loads
#define LN2     0.69314718055994531f
#define INVLN2  1.4426950408889634f

union AFrag { v16h v; uint4    q[2];  };
union BFrag { v16h v; _Float16 h[16]; };
union CFrag { v8f  v; float    f[8];  };

__global__ __launch_bounds__(128) void crf_scan(
    const float* __restrict__ em,       // [B, T, L]
    const int*   __restrict__ seq_lens, // [B]
    const float* __restrict__ trans,    // [L, L]
    float*       __restrict__ out)      // alpha [B,T,L] then logZ [B]
{
  __shared__ __align__(16) _Float16 sV[ROWS * VPITCH];
  __shared__ __align__(16) float    sPartT[ROWS][4];   // [row][wave] maxima
  __shared__ float sLast[ROWS * LPITCH];
  __shared__ float sDump[32];

  const int tid   = threadIdx.x;
  const int wave  = tid >> 5;
  const int lane  = tid & 31;
  const int bbase = blockIdx.x * ROWS;

  const int  colL   = lane & 15;
  const int  col    = wave * 16 + colL;       // label column this lane owns
  const bool hiHalf = (lane >= 16);
  const int  rbase  = hiHalf ? 8 : 0;         // D rows this lane holds
  const int  koffB  = hiHalf ? 16 : 0;        // B-layout K offset
  const int  koffA  = hiHalf ? 16 : 0;        // A-layout K offset (halves)

  auto gidx = [&](int row, int t) {
    return ((size_t)(bbase + row) * CRF_T + (size_t)t) * CRF_L + col;
  };

  // B fragments: exp(transition) columns for this wave's N-tile, built once
  // from global and resident in VGPRs for the entire 511-step scan.
  // B 32x16, this code's contiguous K order (same as A): N = lane&15; lanes 0-15 K=0..15, 16-31 K=16..31.
  BFrag b0, b1;
  #pragma unroll
  for (int h = 0; h < 16; ++h) {
    b0.h[h] = (_Float16)__builtin_amdgcn_exp2f(
                  trans[(koffB + h) * CRF_L + col] * INVLN2);
    b1.h[h] = (_Float16)__builtin_amdgcn_exp2f(
                  trans[(32 + koffB + h) * CRF_L + col] * INVLN2);
  }

  int lenr[8];
  #pragma unroll
  for (int k = 0; k < 8; ++k) lenr[k] = seq_lens[bbase + rbase + k];

  float mrow[8];   // running per-row max, kept in registers across steps

  // ---- t = 0: alpha0 = emissions[:,0,:] ----
  {
    float a[8];
    #pragma unroll
    for (int k = 0; k < 8; ++k) {
      a[k] = em[gidx(rbase + k, 0)];
      out[gidx(rbase + k, 0)] = a[k];
    }
    #pragma unroll
    for (int k = 0; k < 8; ++k) {
      float pm = a[k];
      pm = fmaxf(pm, __shfl_xor(pm, 1, 16));
      pm = fmaxf(pm, __shfl_xor(pm, 2, 16));
      pm = fmaxf(pm, __shfl_xor(pm, 4, 16));
      pm = fmaxf(pm, __shfl_xor(pm, 8, 16));
      if (colL == 0) sPartT[rbase + k][wave] = pm;
    }
    __syncthreads();
    #pragma unroll
    for (int k = 0; k < 8; ++k) {
      float4 p = *reinterpret_cast<const float4*>(&sPartT[rbase + k][0]);
      mrow[k] = fmaxf(fmaxf(p.x, p.y), fmaxf(p.z, p.w));
      sV[(rbase + k) * VPITCH + col] =
          (_Float16)__builtin_amdgcn_exp2f((a[k] - mrow[k]) * INVLN2);
      float* cap = (lenr[k] == 1) ? &sLast[(rbase + k) * LPITCH + col]
                                  : &sDump[lane];
      *cap = a[k];
    }
    __syncthreads();
  }

  // Software-pipelined emission prefetch.
  float emc[8];
  #pragma unroll
  for (int k = 0; k < 8; ++k) emc[k] = em[gidx(rbase + k, 1)];

  const int Mrow = colL;  // A-layout: M = lane&15

  for (int t = 1; t < CRF_T; ++t) {
    // Prefetch next step's emissions; overlaps WMMA + reductions below.
    const int tn = (t < CRF_T - 1) ? t + 1 : t;
    float emn[8];
    #pragma unroll
    for (int k = 0; k < 8; ++k) emn[k] = em[gidx(rbase + k, tn)];

    // A fragments: v (f16) 16x64, K split 0..31 / 32..63.
    // A 16x32, this code's contiguous K order (same as B): M = lane&15; lanes 0-15 K 0..15, 16-31 K 16..31.
    AFrag a0, a1;
    const _Float16* vrow = &sV[Mrow * VPITCH];
    a0.q[0] = *reinterpret_cast<const uint4*>(vrow + koffA);
    a0.q[1] = *reinterpret_cast<const uint4*>(vrow + koffA + 8);
    a1.q[0] = *reinterpret_cast<const uint4*>(vrow + koffA + 32);
    a1.q[1] = *reinterpret_cast<const uint4*>(vrow + koffA + 40);

    CFrag c{};
    c.v = __builtin_amdgcn_wmma_f32_16x16x32_f16(false, a0.v, false, b0.v,
                                                 (short)0, c.v, false, false);
    c.v = __builtin_amdgcn_wmma_f32_16x16x32_f16(false, a1.v, false, b1.v,
                                                 (short)0, c.v, false, false);

    // alpha_new = m + log(dot) + em.  Raw v_log_f32 is safe: dot is far
    // from denormal/inf by construction (v<=1, expT bounded).
    float a[8];
    #pragma unroll
    for (int k = 0; k < 8; ++k) {
      a[k] = mrow[k] + __builtin_amdgcn_logf(c.f[k]) * LN2 + emc[k];
      out[gidx(rbase + k, t)] = a[k];
      // Branchless capture of alpha at t == len-1 (cndmask'd address).
      float* cap = (t == lenr[k] - 1) ? &sLast[(rbase + k) * LPITCH + col]
                                      : &sDump[lane];
      *cap = a[k];
    }

    // Per-row max over 64 labels: shfl_xor tree within each 16-lane half
    // (16 cols), then combine the 4 wave partials through a 16x4 LDS tile.
    #pragma unroll
    for (int k = 0; k < 8; ++k) {
      float pm = a[k];
      pm = fmaxf(pm, __shfl_xor(pm, 1, 16));
      pm = fmaxf(pm, __shfl_xor(pm, 2, 16));
      pm = fmaxf(pm, __shfl_xor(pm, 4, 16));
      pm = fmaxf(pm, __shfl_xor(pm, 8, 16));
      if (colL == 0) sPartT[rbase + k][wave] = pm;
    }
    __syncthreads();

    #pragma unroll
    for (int k = 0; k < 8; ++k) {
      float4 p = *reinterpret_cast<const float4*>(&sPartT[rbase + k][0]);
      mrow[k] = fmaxf(fmaxf(p.x, p.y), fmaxf(p.z, p.w));
      sV[(rbase + k) * VPITCH + col] =
          (_Float16)__builtin_amdgcn_exp2f((a[k] - mrow[k]) * INVLN2);
    }
    __syncthreads();

    #pragma unroll
    for (int k = 0; k < 8; ++k) emc[k] = emn[k];
  }

  // log_Z from the alpha row captured at t == len-1.
  if (tid < ROWS) {
    float mx = -INFINITY;
    for (int j = 0; j < CRF_L; ++j) mx = fmaxf(mx, sLast[tid * LPITCH + j]);
    float s = 0.f;
    for (int j = 0; j < CRF_L; ++j)
      s += __builtin_amdgcn_exp2f((sLast[tid * LPITCH + j] - mx) * INVLN2);
    out[(size_t)CRF_B * CRF_T * CRF_L + bbase + tid] =
        mx + __builtin_amdgcn_logf(s) * LN2;
  }
}

extern "C" void kernel_launch(void* const* d_in, const int* in_sizes, int n_in,
                              void* d_out, int out_size, void* d_ws, size_t ws_size,
                              hipStream_t stream) {
  const float* em    = (const float*)d_in[0];  // [32, 512, 64] f32
  const int*   lens  = (const int*)  d_in[1];  // [32] i32
  const float* trans = (const float*)d_in[2];  // [64, 64] f32
  float* out = (float*)d_out;                  // alpha flat + logZ

  crf_scan<<<CRF_B / ROWS, 128, 0, stream>>>(em, lens, trans, out);
}
